// RWKV_Tmix_x060_38560216384228
// MI455X (gfx1250) — compile-verified, matrix operand order checked by reading, not run on hardware
//
#include <hip/hip_runtime.h>
#include <hip/hip_bf16.h>
#include <math.h>

// ---------------- problem constants ----------------
static constexpr int kB  = 4;
static constexpr int kT  = 512;
static constexpr int kC  = 2048;
static constexpr int kH  = 32;
static constexpr int kN  = 64;
static constexpr int kBT = kB * kT;            // 2048 rows
static constexpr int kTMX = 32;                // TIME_MIX_EXTRA_DIM
static constexpr int kLORA = 5 * kTMX;         // 160
static constexpr int kTDX = 64;                // TIME_DECAY_EXTRA_DIM
static constexpr float kGN_EPS = 1e-5f * 64.0f;

// ---------------- WMMA types ----------------
typedef __bf16  v16bf __attribute__((ext_vector_type(16)));
typedef float   v8f   __attribute__((ext_vector_type(8)));
union AF  { unsigned u[8]; v16bf v; };
union BQ  { uint4 q[2]; __bf16 h[16]; };

#define ACT_NONE 0
#define ACT_TANH 1
#define ACT_SILU 2

// ---------------- f32 -> bf16 convert ----------------
__global__ void k_cvt_bf16(const float* __restrict__ src, __bf16* __restrict__ dst, int n) {
    int i = blockIdx.x * 256 + threadIdx.x;
    if (i < n) dst[i] = (__bf16)src[i];
}

// ---------------- block reduction helper (blockDim = 256) ----------------
__device__ __forceinline__ float block_reduce_sum256(float v, float* red) {
    int t = threadIdx.x;
    red[t] = v; __syncthreads();
    for (int s = 128; s > 0; s >>= 1) {
        if (t < s) red[t] += red[t + s];
        __syncthreads();
    }
    float r = red[0];
    __syncthreads();
    return r;
}

// ---------------- StateEncoder: s0[h][j][i], one block per head ----------------
__global__ __launch_bounds__(256) void k_state_encoder(
    const float* __restrict__ enc_state, const float* __restrict__ sp1_w,
    const float* __restrict__ sp1_b, const float* __restrict__ ln1_g,
    const float* __restrict__ ln1_b, const float* __restrict__ sp2_w,
    const float* __restrict__ sp2_b, const float* __restrict__ ln2_g,
    const float* __restrict__ ln2_b, float* __restrict__ s0) {
    const int h = blockIdx.x, t = threadIdx.x;
    __shared__ float enc[64];
    __shared__ float hid[512];
    __shared__ float red[256];
    if (t < 64) enc[t] = enc_state[h * 64 + t];
    __syncthreads();
    for (int j = t; j < 512; j += 256) {
        float a = sp1_b[j];
        #pragma unroll 8
        for (int e = 0; e < 64; ++e) a = fmaf(enc[e], sp1_w[e * 512 + j], a);
        hid[j] = a;
    }
    __syncthreads();
    float p0 = hid[t], p1 = hid[t + 256];
    float mu = block_reduce_sum256(p0 + p1, red) * (1.f / 512.f);
    float d0 = p0 - mu, d1 = p1 - mu;
    float var = block_reduce_sum256(d0 * d0 + d1 * d1, red) * (1.f / 512.f);
    float inv = rsqrtf(var + 1e-5f);
    hid[t]       = tanhf(d0 * inv * ln1_g[t]       + ln1_b[t]);
    hid[t + 256] = tanhf(d1 * inv * ln1_g[t + 256] + ln1_b[t + 256]);
    __syncthreads();
    float vals[16];
    float lsum = 0.f;
    for (int it = 0; it < 16; ++it) {
        int j = it * 256 + t;
        float a = sp2_b[j];
        #pragma unroll 4
        for (int d = 0; d < 512; ++d) a = fmaf(hid[d], sp2_w[d * 4096 + j], a);
        vals[it] = a; lsum += a;
    }
    float mu2 = block_reduce_sum256(lsum, red) * (1.f / 4096.f);
    float lss = 0.f;
    #pragma unroll
    for (int it = 0; it < 16; ++it) { float dd = vals[it] - mu2; lss += dd * dd; }
    float var2 = block_reduce_sum256(lss, red) * (1.f / 4096.f);
    float inv2 = rsqrtf(var2 + 1e-5f);
    for (int it = 0; it < 16; ++it) {
        int j = it * 256 + t;
        s0[h * 4096 + j] = (vals[it] - mu2) * inv2 * ln2_g[j] + ln2_b[j];
    }
}

// ---------------- token shift ----------------
__global__ void k_prep(const float* __restrict__ x, const float* __restrict__ tss,
                       const float* __restrict__ tmx, float* __restrict__ xx,
                       __bf16* __restrict__ xxxb) {
    int idx = blockIdx.x * 256 + threadIdx.x;
    int c = idx % kC;
    int t = (idx / kC) % kT;
    float xv = x[idx];
    float prev = (t == 0) ? tss[c] : x[idx - kC];
    float d = prev - xv;
    xx[idx] = d;
    xxxb[idx] = (__bf16)(xv + d * tmx[c]);
}

// ---------------- bf16 WMMA GEMM: out(M x Nc) = act(A(M x K) @ W(K x Nc)) ----------
// block tile 128x128, 8 waves in 4x2, each wave 32x64 -> acc[2][4]
template <int ACT>
__global__ __launch_bounds__(256) void k_gemm_bf16(
    const __bf16* __restrict__ A, const __bf16* __restrict__ W,
    float* __restrict__ out, int M, int Ncols, int K) {
    __shared__ __bf16 As[128 * 40];   // 128 rows x 32 k (stride 40)
    __shared__ __bf16 Bs[128 * 40];   // 128 cols x 32 k (stride 40, transposed)
    const int tid  = threadIdx.x;
    const int lane = tid & 31, wid = tid >> 5;
    const int wm = wid & 3, wn = wid >> 2;            // 4x2 wave grid
    const int rowBase = blockIdx.y * 128;
    const int colBase = blockIdx.x * 128;
    const int lrow = lane & 15, khalf = lane >> 4;
    int kofs[8];
    #pragma unroll
    for (int p = 0; p < 8; ++p)                        // ISA 16-bit A/B K striping
        kofs[p] = ((p < 4) ? 2 * p : 16 + 2 * (p - 4)) + khalf * 8;
    v8f acc[2][4] = {};
    const int arow = tid >> 1, ahalf = tid & 1;        // A tile: 2 threads / row
    const int bkr = tid >> 3, bn0 = (tid & 7) * 16;    // B tile: 16 cols / thread

    auto compute = [&]() {
        AF bfrag[4];
        #pragma unroll
        for (int sn = 0; sn < 4; ++sn)
            #pragma unroll
            for (int p = 0; p < 8; ++p)
                bfrag[sn].u[p] =
                    *(const unsigned*)&Bs[(wn * 64 + sn * 16 + lrow) * 40 + kofs[p]];
        #pragma unroll
        for (int sm = 0; sm < 2; ++sm) {
            AF a;
            #pragma unroll
            for (int p = 0; p < 8; ++p)
                a.u[p] = *(const unsigned*)&As[(wm * 32 + sm * 16 + lrow) * 40 + kofs[p]];
            #pragma unroll
            for (int sn = 0; sn < 4; ++sn)
                acc[sm][sn] = __builtin_amdgcn_wmma_f32_16x16x32_bf16(
                    false, a.v, false, bfrag[sn].v, (short)0, acc[sm][sn], false, false);
        }
    };
    auto act_apply = [](float v) -> float {
        if (ACT == ACT_TANH) return tanhf(v);
        if (ACT == ACT_SILU) return v / (1.f + __expf(-v));
        return v;
    };
    const int rsel = khalf * 8;
    const bool full = (colBase + 128 <= Ncols);        // block-uniform

    if (full) {
        for (int kb = 0; kb < K; kb += 32) {
            const __bf16* asrc = A + (size_t)(rowBase + arow) * K + kb + ahalf * 16;
            uint4 a0 = *(const uint4*)asrc;
            uint4 a1 = *(const uint4*)(asrc + 8);
            *(uint4*)&As[arow * 40 + ahalf * 16]     = a0;
            *(uint4*)&As[arow * 40 + ahalf * 16 + 8] = a1;
            const __bf16* wsrc = W + (size_t)(kb + bkr) * Ncols + colBase + bn0;
            BQ b;
            b.q[0] = *(const uint4*)wsrc;
            b.q[1] = *(const uint4*)(wsrc + 8);
            #pragma unroll
            for (int j = 0; j < 16; ++j)
                Bs[(bn0 + j) * 40 + bkr] = b.h[j];
            if (kb + 32 < K) {
                __builtin_prefetch(asrc + 32, 0, 1);
                __builtin_prefetch(wsrc + (size_t)32 * Ncols, 0, 1);
            }
            __syncthreads();
            compute();
            __syncthreads();
        }
        #pragma unroll
        for (int sm = 0; sm < 2; ++sm)
            #pragma unroll
            for (int sn = 0; sn < 4; ++sn) {
                int col = colBase + wn * 64 + sn * 16 + lrow;
                #pragma unroll
                for (int g = 0; g < 8; ++g) {
                    int row = rowBase + wm * 32 + sm * 16 + g + rsel;
                    out[(size_t)row * Ncols + col] = act_apply(acc[sm][sn][g]);
                }
            }
    } else {
        for (int kb = 0; kb < K; kb += 32) {
            const __bf16* asrc = A + (size_t)(rowBase + arow) * K + kb + ahalf * 16;
            uint4 a0 = *(const uint4*)asrc;
            uint4 a1 = *(const uint4*)(asrc + 8);
            *(uint4*)&As[arow * 40 + ahalf * 16]     = a0;
            *(uint4*)&As[arow * 40 + ahalf * 16 + 8] = a1;
            const __bf16* wsrc = W + (size_t)(kb + bkr) * Ncols + colBase + bn0;
            #pragma unroll
            for (int j = 0; j < 16; ++j) {
                __bf16 v = (__bf16)0.f;
                if (colBase + bn0 + j < Ncols) v = wsrc[j];
                Bs[(bn0 + j) * 40 + bkr] = v;
            }
            __syncthreads();
            compute();
            __syncthreads();
        }
        #pragma unroll
        for (int sm = 0; sm < 2; ++sm)
            #pragma unroll
            for (int sn = 0; sn < 4; ++sn) {
                int col = colBase + wn * 64 + sn * 16 + lrow;
                if (col >= Ncols) continue;
                #pragma unroll
                for (int g = 0; g < 8; ++g) {
                    int row = rowBase + wm * 32 + sm * 16 + g + rsel;
                    out[(size_t)row * Ncols + col] = act_apply(acc[sm][sn][g]);
                }
            }
    }
}

// ---------------- 5-way LoRA mix ----------------
struct TM5 { const float* p[5]; };
__global__ void k_mix5(const float* __restrict__ x, const float* __restrict__ xx,
                       const float* __restrict__ m, const float* __restrict__ w2,
                       TM5 tm, __bf16* __restrict__ xmix) {
    int z = blockIdx.z;
    int idx = blockIdx.x * 256 + threadIdx.x;
    int row = idx / kC, c = idx % kC;
    const float* mrow = m + (size_t)row * kLORA + z * kTMX;
    const float* wz   = w2 + ((size_t)z * kTMX) * kC + c;
    float acc = 0.f;
    #pragma unroll
    for (int d = 0; d < kTMX; ++d) acc = fmaf(mrow[d], wz[(size_t)d * kC], acc);
    float val = tm.p[z][c] + acc;
    xmix[(size_t)z * kBT * kC + idx] = (__bf16)(x[idx] + xx[idx] * val);
}

// ---------------- decay: d = exp(-exp(time_decay + tanh(xw@w1)@w2)) ----------------
__global__ void k_decay2(const float* __restrict__ tdec, const float* __restrict__ dec_w2,
                         const float* __restrict__ time_decay, float* __restrict__ dout) {
    int idx = blockIdx.x * 256 + threadIdx.x;
    int row = idx / kC, c = idx % kC;
    const float* tr = tdec + (size_t)row * kTDX;
    float acc = 0.f;
    #pragma unroll
    for (int j = 0; j < kTDX; ++j) acc = fmaf(tr[j], dec_w2[(size_t)j * kC + c], acc);
    float w = time_decay[c] + acc;
    dout[idx] = __expf(-__expf(w));
}

// ---------------- WKV6 scan: one (b,h) per 64-thread block, state in VGPRs --------
__global__ __launch_bounds__(64) void k_wkv(
    const float* __restrict__ r, const float* __restrict__ k,
    const float* __restrict__ v, const float* __restrict__ d,
    const float* __restrict__ s0, const float* __restrict__ u_,
    float* __restrict__ o) {
    int bh = blockIdx.x;
    int b = bh / kH, h = bh % kH;
    int i = threadIdx.x;
    __shared__ float su[64], sr[64], sk[64], sv[64], sd[64];
    su[i] = u_[h * 64 + i];
    float s[64];
    #pragma unroll
    for (int j = 0; j < 64; ++j) s[j] = s0[h * 4096 + j * 64 + i];
    __syncthreads();
    const size_t base0 = (size_t)b * kT * kC + h * 64;
    for (int t = 0; t < kT; ++t) {
        size_t base = base0 + (size_t)t * kC;
        sr[i] = r[base + i]; sk[i] = k[base + i];
        sv[i] = v[base + i]; sd[i] = d[base + i];
        __syncthreads();
        float vi = sv[i];
        float y = 0.f;
        #pragma unroll
        for (int j = 0; j < 64; ++j) {
            float kv = sk[j] * vi;
            y += sr[j] * fmaf(su[j], kv, s[j]);
            s[j] = fmaf(s[j], sd[j], kv);
        }
        o[base + i] = y;
        __syncthreads();
    }
}

// ---------------- GroupNorm(H) + scale/bias + gate -> bf16 -------------------------
__global__ __launch_bounds__(64) void k_gnorm(
    const float* __restrict__ o, const float* __restrict__ g,
    const float* __restrict__ gn_g, const float* __restrict__ gn_b,
    __bf16* __restrict__ gated) {
    int grp = blockIdx.x;
    int bt = grp >> 5, h = grp & 31;
    int i = threadIdx.x;
    __shared__ float sh[64];
    size_t base = (size_t)bt * kC + h * 64;
    float v = o[base + i];
    sh[i] = v; __syncthreads();
    for (int s = 32; s > 0; s >>= 1) { if (i < s) sh[i] += sh[i + s]; __syncthreads(); }
    float mu = sh[0] * (1.f / 64.f); __syncthreads();
    float dv = v - mu;
    sh[i] = dv * dv; __syncthreads();
    for (int s = 32; s > 0; s >>= 1) { if (i < s) sh[i] += sh[i + s]; __syncthreads(); }
    float var = sh[0] * (1.f / 64.f);
    float inv = rsqrtf(var + kGN_EPS);
    int c = h * 64 + i;
    float gn = dv * inv * gn_g[c] + gn_b[c];
    gated[base + i] = (__bf16)(gn * g[base + i]);
}

// =============================== launch ===============================
extern "C" void kernel_launch(void* const* d_in, const int* in_sizes, int n_in,
                              void* d_out, int out_size, void* d_ws, size_t ws_size,
                              hipStream_t stream) {
    (void)in_sizes; (void)n_in; (void)out_size; (void)ws_size;
    const float* x          = (const float*)d_in[0];
    const float* time_maa_x = (const float*)d_in[1];
    const float* time_maa_w = (const float*)d_in[2];
    const float* time_maa_k = (const float*)d_in[3];
    const float* time_maa_v = (const float*)d_in[4];
    const float* time_maa_r = (const float*)d_in[5];
    const float* time_maa_g = (const float*)d_in[6];
    const float* maa_w1     = (const float*)d_in[7];
    const float* maa_w2     = (const float*)d_in[8];
    const float* time_decay = (const float*)d_in[9];
    const float* dec_w1     = (const float*)d_in[10];
    const float* dec_w2     = (const float*)d_in[11];
    const float* time_faaaa = (const float*)d_in[12];
    const float* tss        = (const float*)d_in[13];
    const float* Wr         = (const float*)d_in[14];
    const float* Wk         = (const float*)d_in[15];
    const float* Wv         = (const float*)d_in[16];
    const float* Wg         = (const float*)d_in[17];
    const float* Wo         = (const float*)d_in[18];
    const float* gn_g       = (const float*)d_in[19];
    const float* gn_b       = (const float*)d_in[20];
    const float* enc_state  = (const float*)d_in[21];
    const float* sp1_w      = (const float*)d_in[22];
    const float* sp1_b      = (const float*)d_in[23];
    const float* ln1_g      = (const float*)d_in[24];
    const float* ln1_b      = (const float*)d_in[25];
    const float* sp2_w      = (const float*)d_in[26];
    const float* sp2_b      = (const float*)d_in[27];
    const float* ln2_g      = (const float*)d_in[28];
    const float* ln2_b      = (const float*)d_in[29];
    float* out = (float*)d_out;

    char* ws = (char*)d_ws;
    size_t off = 0;
    auto ws_alloc = [&](size_t bytes) -> void* {
        void* p = (void*)(ws + off);
        off += (bytes + 255) & ~(size_t)255;
        return p;
    };
    const size_t CC = (size_t)kC * kC, BTC = (size_t)kBT * kC;
    __bf16* wWr   = (__bf16*)ws_alloc(CC * 2);
    __bf16* wWk   = (__bf16*)ws_alloc(CC * 2);
    __bf16* wWv   = (__bf16*)ws_alloc(CC * 2);
    __bf16* wWg   = (__bf16*)ws_alloc(CC * 2);
    __bf16* wWo   = (__bf16*)ws_alloc(CC * 2);
    __bf16* wMaa1 = (__bf16*)ws_alloc((size_t)kC * kLORA * 2);
    __bf16* wDec1 = (__bf16*)ws_alloc((size_t)kC * kTDX * 2);
    float*  s0    = (float*)ws_alloc((size_t)kH * kN * kN * 4);
    float*  xxbuf = (float*)ws_alloc(BTC * 4);
    __bf16* xxxbf = (__bf16*)ws_alloc(BTC * 2);
    float*  mbuf  = (float*)ws_alloc((size_t)kBT * kLORA * 4);
    __bf16* xmix  = (__bf16*)ws_alloc(5 * BTC * 2);
    float*  rbuf  = (float*)ws_alloc(BTC * 4);
    float*  kbuf  = (float*)ws_alloc(BTC * 4);
    float*  vbuf  = (float*)ws_alloc(BTC * 4);
    float*  gbuf  = (float*)ws_alloc(BTC * 4);
    float*  tdec  = (float*)ws_alloc((size_t)kBT * kTDX * 4);
    float*  dbuf  = (float*)ws_alloc(BTC * 4);
    float*  obuf  = (float*)ws_alloc(BTC * 4);
    __bf16* gated = (__bf16*)ws_alloc(BTC * 2);

    auto cvt = [&](const float* s, __bf16* d, size_t n) {
        k_cvt_bf16<<<dim3((unsigned)((n + 255) / 256)), 256, 0, stream>>>(s, d, (int)n);
    };
    cvt(Wr, wWr, CC); cvt(Wk, wWk, CC); cvt(Wv, wWv, CC);
    cvt(Wg, wWg, CC); cvt(Wo, wWo, CC);
    cvt(maa_w1, wMaa1, (size_t)kC * kLORA);
    cvt(dec_w1, wDec1, (size_t)kC * kTDX);

    k_state_encoder<<<kH, 256, 0, stream>>>(enc_state, sp1_w, sp1_b, ln1_g, ln1_b,
                                            sp2_w, sp2_b, ln2_g, ln2_b, s0);
    k_prep<<<(unsigned)(BTC / 256), 256, 0, stream>>>(x, tss, time_maa_x, xxbuf, xxxbf);

    // m = tanh(xxx @ maa_w1)   (2048 x 160, K=2048) -> 2 col blocks (1 partial)
    k_gemm_bf16<ACT_TANH><<<dim3((kLORA + 127) / 128, kBT / 128), 256, 0, stream>>>(
        xxxbf, wMaa1, mbuf, kBT, kLORA, kC);

    TM5 tm; tm.p[0] = time_maa_w; tm.p[1] = time_maa_k; tm.p[2] = time_maa_v;
    tm.p[3] = time_maa_r; tm.p[4] = time_maa_g;
    k_mix5<<<dim3((unsigned)(BTC / 256), 1, 5), 256, 0, stream>>>(
        x, xxbuf, mbuf, maa_w2, tm, xmix);

    dim3 gproj(kC / 128, kBT / 128);
    k_gemm_bf16<ACT_NONE><<<gproj, 256, 0, stream>>>(xmix + 3 * BTC, wWr, rbuf, kBT, kC, kC);
    k_gemm_bf16<ACT_NONE><<<gproj, 256, 0, stream>>>(xmix + 1 * BTC, wWk, kbuf, kBT, kC, kC);
    k_gemm_bf16<ACT_NONE><<<gproj, 256, 0, stream>>>(xmix + 2 * BTC, wWv, vbuf, kBT, kC, kC);
    k_gemm_bf16<ACT_SILU><<<gproj, 256, 0, stream>>>(xmix + 4 * BTC, wWg, gbuf, kBT, kC, kC);

    k_gemm_bf16<ACT_TANH><<<dim3(1, kBT / 128), 256, 0, stream>>>(
        xmix + 0 * BTC, wDec1, tdec, kBT, kTDX, kC);
    k_decay2<<<(unsigned)(BTC / 256), 256, 0, stream>>>(tdec, dec_w2, time_decay, dbuf);

    k_wkv<<<kB * kH, 64, 0, stream>>>(rbuf, kbuf, vbuf, dbuf, s0, time_faaaa, obuf);
    k_gnorm<<<(unsigned)((size_t)kBT * kH), 64, 0, stream>>>(obuf, gbuf, gn_g, gn_b, gated);
    k_gemm_bf16<ACT_NONE><<<gproj, 256, 0, stream>>>(gated, wWo, out, kBT, kC, kC);
}
